// GATModel_85907935854812
// MI455X (gfx1250) — matrix operand order repaired and checked by reading, not compiled, not run on hardware
//
#include <hip/hip_runtime.h>
#include <hip/hip_bf16.h>

// ---------------------------------------------------------------------------
// GAT layer for MI455X (gfx1250, wave32, WMMA).
// Sizes: N=20000 nodes, E=320000 edges (+N self loops), IN=128, H=8, C=120.
// ---------------------------------------------------------------------------

typedef __attribute__((ext_vector_type(16))) _Float16 v16h;
typedef __attribute__((ext_vector_type(8)))  float    v8f;

#define HC    960
#define INF_  128
#define HEADS 8
#define CHAN  120
#define SLOPE 0.2f
#define LROW  24   // padded per-lane fragment row (halfs): 48B, 16B-aligned

// ---------------------------------------------------------------------------
// K1: xp = x @ W  via v_wmma_f32_16x16x32_f16.
// grid = (N/16, 8); block = 256 (8 waves). Each block: one 16-row M tile,
// 128 columns of N (one 16-col tile per wave). K=128 -> 4 steps of 32.
//
// LDS tiles are staged in *fragment order*: AshF[kb][lane][0..15] is exactly
// the 16 halves lane needs for k-step kb -> fragment load = 2x ds_load_b128.
//
// 16-bit A 16x32 lane layout:  lane = half*16 + m,
//   elem (2v+par) holds A[m][ (v<4?0:16) + half*8 + (v&3)*2 + par ]
// 16-bit B 32x16 lane layout:  lane = half*16 + n,
//   elem e holds B[ (e<8?0:16) + half*8 + (e&7) ][ n ]
// ---------------------------------------------------------------------------
__global__ __launch_bounds__(256) void gemm_wmma_kernel(
    const float* __restrict__ X, const float* __restrict__ W,
    float* __restrict__ XP, int N)
{
    __shared__ __align__(16) _Float16 AshF[4][32][LROW];        //  6 KB
    __shared__ __align__(16) _Float16 BshF[8][4][32][LROW];     // 48 KB

    const int tid   = threadIdx.x;
    const int mbase = blockIdx.x * 16;
    const int nblk  = blockIdx.y * 128;

    // ---- stage A tile (16x128 fp32 -> f16, fragment order) ----
    // 512 float4 loads, 256 threads -> 2 each
    for (int i4 = tid; i4 < 16 * 32; i4 += 256) {
        int r  = i4 >> 5;
        int c4 = (i4 & 31) * 4;
        int gm = mbase + r;
        float4 v = {0.f, 0.f, 0.f, 0.f};
        if (gm < N) v = *(const float4*)(X + (size_t)gm * INF_ + c4);
        #pragma unroll
        for (int e = 0; e < 4; e++) {
            int c    = c4 + e;
            int kb   = c >> 5, kk = c & 31;
            int par  = kk & 1, t = kk >> 1;            // t in 0..15
            int vv   = ((t >> 3) << 2) | (t & 3);      // VGPR 0..7
            int half = (t >> 2) & 1;
            int lane = half * 16 + r;
            float fv = (&v.x)[e];
            AshF[kb][lane][2 * vv + par] = (_Float16)fv;
        }
    }
    // ---- stage B tile (128 x up-to-128 fp32 -> f16, fragment order) ----
    int ncols = HC - nblk; if (ncols > 128) ncols = 128;   // 128 or 64, 4-aligned
    for (int i4 = tid; i4 < 128 * 32; i4 += 256) {
        int k  = i4 >> 5;
        int c4 = (i4 & 31) * 4;
        float4 v = {0.f, 0.f, 0.f, 0.f};
        if (c4 < ncols) v = *(const float4*)(W + (size_t)k * HC + nblk + c4);
        #pragma unroll
        for (int e = 0; e < 4; e++) {
            int c    = c4 + e;
            int wt   = c >> 4, n = c & 15;
            int kb   = k >> 5, kk = k & 31;
            int half = (kk >> 3) & 1, rem = (kk & 7) | ((kk >> 4) << 3);
            int lane = half * 16 + n;
            float fv = (&v.x)[e];
            BshF[wt][kb][lane][rem] = (_Float16)fv;
        }
    }
    __syncthreads();

    const int wave = tid >> 5;
    const int lane = tid & 31;
    const int ncol0 = wave * 16;
    if (nblk + ncol0 >= HC) return;            // wave-uniform: EXEC stays full

    v8f acc = {};
    #pragma unroll
    for (int kb = 0; kb < 4; kb++) {
        v16h a = *(const v16h*)(&AshF[kb][lane][0]);        // 2x ds_load_b128
        v16h b = *(const v16h*)(&BshF[wave][kb][lane][0]);  // 2x ds_load_b128
        acc = __builtin_amdgcn_wmma_f32_16x16x32_f16(
            false, a, false, b, (short)0, acc, false, false);
    }

    // D layout: VGPR r, lane L -> m = r + (L/16)*8, n = L%16
    const int half = lane >> 4;
    const int gn   = nblk + ncol0 + (lane & 15);
    float* outp = XP + (size_t)(mbase + half * 8) * HC + gn;
    if (mbase + 16 <= N) {
        // full tile (always taken for N % 16 == 0): unguarded coalesced stores
        #pragma unroll
        for (int r = 0; r < 8; r++)
            outp[(size_t)r * HC] = acc[r];
    } else {
        #pragma unroll
        for (int r = 0; r < 8; r++) {
            int gm = mbase + r + half * 8;
            if (gm < N) XP[(size_t)gm * HC + gn] = acc[r];
        }
    }
}

// ---------------------------------------------------------------------------
// K2: per-(node,head) attention logits a_s, a_d (120-length dots).
// One wave per (node,head); 8 waves per 256-thread block; grid = N blocks.
// ---------------------------------------------------------------------------
__global__ __launch_bounds__(256) void att_kernel(
    const float* __restrict__ xp, const float* __restrict__ att_src,
    const float* __restrict__ att_dst, float* __restrict__ a_s,
    float* __restrict__ a_d, int N)
{
    int w    = (blockIdx.x * 256 + threadIdx.x) >> 5;
    int lane = threadIdx.x & 31;
    if (w >= N * HEADS) return;
    int n = w >> 3, h = w & 7;
    const float* xrow = xp + (size_t)n * HC + h * CHAN;
    const float* as   = att_src + h * CHAN;
    const float* ad   = att_dst + h * CHAN;
    float s0 = 0.f, s1 = 0.f;
    for (int c = lane; c < CHAN; c += 32) {
        float v = xrow[c];
        s0 += v * as[c];
        s1 += v * ad[c];
    }
    #pragma unroll
    for (int off = 16; off; off >>= 1) {
        s0 += __shfl_xor(s0, off, 32);
        s1 += __shfl_xor(s1, off, 32);
    }
    if (lane == 0) { a_s[w] = s0; a_d[w] = s1; }
}

// ---------------------------------------------------------------------------
// K3: in-degree histogram over real edges.
// ---------------------------------------------------------------------------
__global__ void degree_kernel(const int* __restrict__ dst, int* __restrict__ deg, int E)
{
    int i = blockIdx.x * blockDim.x + threadIdx.x;
    if (i < E) atomicAdd(&deg[dst[i]], 1);
}

// ---------------------------------------------------------------------------
// K4: single-block exclusive scan of deg -> rowptr[0..N].
// ---------------------------------------------------------------------------
__global__ __launch_bounds__(1024) void scan_kernel(
    const int* __restrict__ deg, int* __restrict__ rowptr, int n)
{
    __shared__ int sh[1024];
    const int tid = threadIdx.x;
    int carry = 0;
    for (int base = 0; base < n; base += 1024) {
        int idx = base + tid;
        int v = (idx < n) ? deg[idx] : 0;
        sh[tid] = v;
        __syncthreads();
        for (int off = 1; off < 1024; off <<= 1) {
            int t = (tid >= off) ? sh[tid - off] : 0;
            __syncthreads();
            sh[tid] += t;
            __syncthreads();
        }
        if (idx < n) rowptr[idx] = carry + sh[tid] - v;   // exclusive
        int tot = sh[1023];
        __syncthreads();
        carry += tot;
    }
    if (tid == 0) rowptr[n] = carry;
}

// ---------------------------------------------------------------------------
// K5: CSR fill: col[] holds src ids grouped by dst.
// ---------------------------------------------------------------------------
__global__ void fill_kernel(const int* __restrict__ src, const int* __restrict__ dst,
                            const int* __restrict__ rowptr, int* __restrict__ cursor,
                            int* __restrict__ col, int E)
{
    int i = blockIdx.x * blockDim.x + threadIdx.x;
    if (i < E) {
        int d   = dst[i];
        int pos = atomicAdd(&cursor[d], 1);
        col[rowptr[d] + pos] = src[i];
    }
}

// ---------------------------------------------------------------------------
// K6: per-destination softmax + gather-aggregate + bias + final [960->2] linear.
// One 128-thread block per node. Self-loop is logical edge index == deg.
// xp (76.8 MB) is L2-resident -> gather served from L2, single write per node.
// Channels handled as float4 slots (240 slots); head stride 120 is 4-aligned,
// so each slot lies in exactly one head -> global_load_b128 gathers.
// ---------------------------------------------------------------------------
__global__ __launch_bounds__(128) void aggregate_kernel(
    const float* __restrict__ xp, const float* __restrict__ a_s,
    const float* __restrict__ a_d, const int* __restrict__ rowptr,
    const int* __restrict__ col, const float* __restrict__ bias,
    const float* __restrict__ fc_w, const float* __restrict__ fc_b,
    float* __restrict__ out)
{
    const int n   = blockIdx.x;
    const int tid = threadIdx.x;
    const int beg = rowptr[n];
    const int deg = rowptr[n + 1] - beg;
    const int total = deg + 1;                 // + self loop

    __shared__ float ad_sh[HEADS];
    __shared__ float m_sh[HEADS], s_sh[HEADS];
    __shared__ float alpha_sh[16 * HEADS];
    __shared__ int   srcid_sh[16];
    __shared__ float red0[128], red1[128];

    if (tid < HEADS) ad_sh[tid] = a_d[(size_t)n * HEADS + tid];
    __syncthreads();

    // ---- stage 1 (wave 0): per-head max & exp-sum over incoming edges ----
    if (tid < 32) {
        float mh[HEADS];
        #pragma unroll
        for (int h = 0; h < HEADS; h++) mh[h] = -3.0e38f;
        for (int i = tid; i < total; i += 32) {
            int s = (i < deg) ? col[beg + i] : n;
            #pragma unroll
            for (int h = 0; h < HEADS; h++) {
                float e = a_s[(size_t)s * HEADS + h] + ad_sh[h];
                e = (e > 0.f) ? e : SLOPE * e;
                mh[h] = fmaxf(mh[h], e);
            }
        }
        #pragma unroll
        for (int h = 0; h < HEADS; h++)
            #pragma unroll
            for (int off = 16; off; off >>= 1)
                mh[h] = fmaxf(mh[h], __shfl_xor(mh[h], off, 32));

        float ss[HEADS];
        #pragma unroll
        for (int h = 0; h < HEADS; h++) ss[h] = 0.f;
        for (int i = tid; i < total; i += 32) {
            int s = (i < deg) ? col[beg + i] : n;
            #pragma unroll
            for (int h = 0; h < HEADS; h++) {
                float e = a_s[(size_t)s * HEADS + h] + ad_sh[h];
                e = (e > 0.f) ? e : SLOPE * e;
                ss[h] += __expf(e - mh[h]);
            }
        }
        #pragma unroll
        for (int h = 0; h < HEADS; h++)
            #pragma unroll
            for (int off = 16; off; off >>= 1)
                ss[h] += __shfl_xor(ss[h], off, 32);

        if (tid == 0) {
            #pragma unroll
            for (int h = 0; h < HEADS; h++) { m_sh[h] = mh[h]; s_sh[h] = ss[h]; }
        }
    }
    __syncthreads();

    // ---- stage 2: weighted float4 gather ----
    const int slot0 = tid;                  // always < 240
    const int slot1 = tid + 128;            // valid if < 240 (tid < 112)
    const int h0 = slot0 / 30;              // head of slot (120 ch = 30 slots)
    const int h1 = (slot1 < 240) ? (slot1 / 30) : 0;
    float4 acc0 = {0.f, 0.f, 0.f, 0.f};
    float4 acc1 = {0.f, 0.f, 0.f, 0.f};

    for (int chunk = 0; chunk < total; chunk += 16) {
        int ne = total - chunk; if (ne > 16) ne = 16;
        // one alpha per (local edge, head): up to 128 values -> one per thread
        if (tid < ne * HEADS) {
            int le = tid >> 3, h = tid & 7;
            int i  = chunk + le;
            int s  = (i < deg) ? col[beg + i] : n;
            if (h == 0) srcid_sh[le] = s;
            float e = a_s[(size_t)s * HEADS + h] + ad_sh[h];
            e = (e > 0.f) ? e : SLOPE * e;
            alpha_sh[le * HEADS + h] = __expf(e - m_sh[h]) / s_sh[h];
        }
        __syncthreads();
        for (int le = 0; le < ne; ++le) {
            const float* xs = xp + (size_t)srcid_sh[le] * HC;
            if (le + 1 < ne)
                __builtin_prefetch(xp + (size_t)srcid_sh[le + 1] * HC + tid * 4, 0, 3);
            float a0 = alpha_sh[le * HEADS + h0];
            float4 v0 = *(const float4*)(xs + 4 * slot0);
            acc0.x += a0 * v0.x; acc0.y += a0 * v0.y;
            acc0.z += a0 * v0.z; acc0.w += a0 * v0.w;
            if (slot1 < 240) {
                float a1 = alpha_sh[le * HEADS + h1];
                float4 v1 = *(const float4*)(xs + 4 * slot1);
                acc1.x += a1 * v1.x; acc1.y += a1 * v1.y;
                acc1.z += a1 * v1.z; acc1.w += a1 * v1.w;
            }
        }
        __syncthreads();
    }

    // ---- stage 3: fused bias + final linear [960 -> 2] ----
    float p0 = 0.f, p1 = 0.f;
    #pragma unroll
    for (int e = 0; e < 4; e++) {
        int ch = 4 * slot0 + e;
        float v = (&acc0.x)[e] + bias[ch];
        p0 += v * fc_w[ch * 2 + 0];
        p1 += v * fc_w[ch * 2 + 1];
    }
    if (slot1 < 240) {
        #pragma unroll
        for (int e = 0; e < 4; e++) {
            int ch = 4 * slot1 + e;
            float v = (&acc1.x)[e] + bias[ch];
            p0 += v * fc_w[ch * 2 + 0];
            p1 += v * fc_w[ch * 2 + 1];
        }
    }
    red0[tid] = p0; red1[tid] = p1;
    __syncthreads();
    for (int off = 64; off; off >>= 1) {
        if (tid < off) { red0[tid] += red0[tid + off]; red1[tid] += red1[tid + off]; }
        __syncthreads();
    }
    if (tid == 0) {
        out[(size_t)n * 2 + 0] = red0[0] + fc_b[0];
        out[(size_t)n * 2 + 1] = red1[0] + fc_b[1];
    }
}

// ---------------------------------------------------------------------------
// Launch
// ---------------------------------------------------------------------------
extern "C" void kernel_launch(void* const* d_in, const int* in_sizes, int n_in,
                              void* d_out, int out_size, void* d_ws, size_t ws_size,
                              hipStream_t stream)
{
    const float* x       = (const float*)d_in[0];
    const int*   ei      = (const int*)d_in[1];
    const float* W       = (const float*)d_in[2];
    const float* att_src = (const float*)d_in[3];
    const float* att_dst = (const float*)d_in[4];
    const float* bias    = (const float*)d_in[5];
    const float* fc_w    = (const float*)d_in[6];
    const float* fc_b    = (const float*)d_in[7];
    float*       out     = (float*)d_out;

    const int N = in_sizes[0] / INF_;   // 20000
    const int E = in_sizes[1] / 2;      // 320000
    const int* src = ei;
    const int* dst = ei + E;

    // workspace layout (256B aligned slices)
    char* base = (char*)d_ws;
    auto alloc = [&](size_t bytes) -> char* {
        char* p = base;
        base += (bytes + 255) & ~(size_t)255;
        return p;
    };
    float* xp     = (float*)alloc((size_t)N * HC * sizeof(float));   // 76.8 MB
    float* a_s    = (float*)alloc((size_t)N * HEADS * sizeof(float));
    float* a_d    = (float*)alloc((size_t)N * HEADS * sizeof(float));
    int*   col    = (int*)alloc((size_t)E * sizeof(int));
    int*   rowptr = (int*)alloc((size_t)(N + 1) * sizeof(int));
    int*   deg    = (int*)alloc((size_t)N * sizeof(int));
    int*   cursor = (int*)alloc((size_t)N * sizeof(int));

    hipMemsetAsync(deg, 0, (size_t)N * sizeof(int), stream);
    hipMemsetAsync(cursor, 0, (size_t)N * sizeof(int), stream);

    dim3 ggrid((N + 15) / 16, (HC + 127) / 128);   // 1250 x 8
    gemm_wmma_kernel<<<ggrid, 256, 0, stream>>>(x, W, xp, N);

    att_kernel<<<N, 256, 0, stream>>>(xp, att_src, att_dst, a_s, a_d, N);

    degree_kernel<<<(E + 255) / 256, 256, 0, stream>>>(dst, deg, E);
    scan_kernel<<<1, 1024, 0, stream>>>(deg, rowptr, N);
    fill_kernel<<<(E + 255) / 256, 256, 0, stream>>>(src, dst, rowptr, cursor, col, E);

    aggregate_kernel<<<N, 128, 0, stream>>>(xp, a_s, a_d, rowptr, col,
                                            bias, fc_w, fc_b, out);
}
